// GNNPathway_89859305767787
// MI455X (gfx1250) — matrix operand order repaired and checked by reading, not compiled, not run on hardware
//
#include <hip/hip_runtime.h>
#include <hip/hip_bf16.h>
#include <math.h>

// ---------------------------------------------------------------------------
// SchNet-style GNN on MI455X (gfx1250, wave32).
// Fused edge pipeline; all GEMMs on v_wmma_f32_16x16x32_bf16 (fp32 accum).
// Weights pre-swizzled in global memory into B-fragment order so every
// fragment load is a contiguous 32B/lane load. A tiles in LDS, loaded as
// ds_load_b128 pairs. Scatter via relaxed agent-scope fp32 atomics.
// ---------------------------------------------------------------------------

#define NNODE 50000
#define NEDGE 1600000
#define NGRAPH 1024
#define HID 128
#define FLT 64
#define NGAUSS 50
#define NLAYER 4
#define RBF_W (10.0f / 49.0f)   // CUTOFF/(NG-1)

typedef __bf16 bf16;
typedef __bf16 v16bf __attribute__((ext_vector_type(16)));
typedef __bf16 v8bf  __attribute__((ext_vector_type(8)));
typedef __bf16 v4bf  __attribute__((ext_vector_type(4)));
typedef __bf16 v2bf  __attribute__((ext_vector_type(2)));
typedef float  v8f   __attribute__((ext_vector_type(8)));
typedef float  v4f   __attribute__((ext_vector_type(4)));

__device__ __forceinline__ float silu_f(float x) {
  return x * __builtin_amdgcn_rcpf(1.0f + __expf(-x));
}
__device__ __forceinline__ bf16 tobf(float x) { return (bf16)x; }

__device__ __forceinline__ void atomic_addf(float* p, float v) {
  __hip_atomic_fetch_add(p, v, __ATOMIC_RELAXED, __HIP_MEMORY_SCOPE_AGENT);
}

// A fragment (16x32 bf16) from row-major LDS tile [16][ld].
// lane L: M=L&15, half=L>>4; needs K = k0+8*half..+7 and k0+16+8*half..+7
// -> two contiguous 16B chunks per lane -> 2x ds_load_b128.
__device__ __forceinline__ v16bf frag_a(const bf16* A, int ld, int k0, int lane) {
  int m = lane & 15, h = lane >> 4;
  const bf16* p = A + m * ld + k0 + (h << 3);
  v8bf lo = *(const v8bf*)p;
  v8bf hi = *(const v8bf*)(p + 16);
  return __builtin_shufflevector(lo, hi, 0, 1, 2, 3, 4, 5, 6, 7,
                                 8, 9, 10, 11, 12, 13, 14, 15);
}

// B fragment from pre-swizzled global weights: fragment f = kc*nT + t is a
// 512-element blob, lane L owns 16 contiguous bf16 (32B) -> global_load_b128 x2.
__device__ __forceinline__ v16bf frag_bg(const bf16* Bsw, int frag, int lane) {
  return *(const v16bf*)(Bsw + ((size_t)frag << 9) + (lane << 4));
}

__device__ __forceinline__ v8f wmma_bf(v16bf a, v16bf b, v8f c) {
  return __builtin_amdgcn_wmma_f32_16x16x32_bf16(false, a, false, b, (short)0, c, false, false);
}

// ---------------------------------------------------------------------------
// fp32 W[K][cols] -> bf16 swizzled B-fragment layout (rows >= srcRows zero pad)
// dst[frag*512 + lane*16 + e] = W[32*kc + 16*(e>>3) + 8*(lane>>4) + (e&7)][16*t + (lane&15)]
// with frag = kc*(cols/16) + t
// ---------------------------------------------------------------------------
__global__ void cvt_swz_kernel(const float* __restrict__ src, bf16* __restrict__ dst,
                               int srcRows, int K, int cols) {
  int i = blockIdx.x * blockDim.x + threadIdx.x;
  if (i >= K * cols) return;
  int frag = i >> 9, r = i & 511;
  int lane = r >> 4, e = r & 15;
  int nT = cols >> 4;
  int kc = frag / nT, t = frag - kc * nT;
  int k = kc * 32 + ((e >> 3) << 4) + ((lane >> 4) << 3) + (e & 7);
  int n = t * 16 + (lane & 15);
  dst[i] = (k < srcRows) ? tobf(src[(size_t)k * cols + n]) : tobf(0.0f);
}

// ---------------------------------------------------------------------------
// embed layer 1: t1[n,c] = silu(z[n,:3] @ W1[:,c] + b1[c])   (K=3 scalar FMA)
// ---------------------------------------------------------------------------
__global__ void emb1_kernel(const float* __restrict__ z, const float* __restrict__ w1,
                            const float* __restrict__ b1, float* __restrict__ t1) {
  int i = blockIdx.x * blockDim.x + threadIdx.x;
  if (i >= NNODE * HID) return;
  int n = i >> 7, c = i & 127;
  float v = b1[c] + z[n * 3 + 0] * w1[0 * HID + c]
                  + z[n * 3 + 1] * w1[1 * HID + c]
                  + z[n * 3 + 2] * w1[2 * HID + c];
  t1[i] = silu_f(v);
}

// ---------------------------------------------------------------------------
// Node GEMM: out[M,128] (=|+=) act(A[M,K] @ W[K,128] + b). K in {64,128}.
// 128 threads (4 waves), 64 rows/block, 16 rows/wave. Fully unrolled WMMA.
// ---------------------------------------------------------------------------
template <int K>
__global__ void gemm_node_kernel(const float* __restrict__ A, int M,
                                 const bf16* __restrict__ Bsw,
                                 const float* __restrict__ bias,
                                 float* __restrict__ out, int act, int addTo) {
  constexpr int NT = 8;          // 128/16
  constexpr int KT = K / 32;
  __shared__ alignas(16) bf16 As[64 * K];
  const int tid = threadIdx.x, lane = tid & 31, w = tid >> 5;
  const int row0 = blockIdx.x * 64;
  const int n = lane & 15, hlf = lane >> 4;

  // stage A (fp32 -> bf16), vectorized 4-wide
  for (int i = tid; i < 64 * (K / 4); i += 128) {
    int r = i / (K / 4), c4 = (i - r * (K / 4)) * 4;
    int row = row0 + r;
    v4bf o;
    if (row < M) {
      v4f f = *(const v4f*)&A[(size_t)row * K + c4];
      o[0] = tobf(f[0]); o[1] = tobf(f[1]); o[2] = tobf(f[2]); o[3] = tobf(f[3]);
    } else {
      o[0] = tobf(0.f); o[1] = tobf(0.f); o[2] = tobf(0.f); o[3] = tobf(0.f);
    }
    *(v4bf*)&As[r * K + c4] = o;
  }
  __syncthreads();

  const bf16* Aw = &As[(w * 16) * K];
  v16bf afr[KT];
#pragma unroll
  for (int kc = 0; kc < KT; ++kc) afr[kc] = frag_a(Aw, K, kc * 32, lane);

  const bool full = (row0 + 64 <= M);
#pragma unroll
  for (int t = 0; t < NT; ++t) {
    v8f c = {};
#pragma unroll
    for (int kc = 0; kc < KT; ++kc) c = wmma_bf(afr[kc], frag_bg(Bsw, kc * NT + t, lane), c);
    float bb = bias[t * 16 + n];
#pragma unroll
    for (int v = 0; v < 8; ++v) {
      int row = row0 + w * 16 + v + 8 * hlf;
      if (full || row < M) {
        float val = c[v] + bb;
        if (act) val = silu_f(val);
        size_t idx = (size_t)row * 128 + t * 16 + n;
        out[idx] = addTo ? (out[idx] + val) : val;
      }
    }
  }
}

// ---------------------------------------------------------------------------
// Fused edge kernel: rbf -> filter MLP (2 GEMMs) -> gather h[col] ->
// project (GEMM) -> msg = hj*W -> atomic scatter into agg[row].
// 128 threads (4 waves), 16 edges/wave. Weights read as swizzled fragments
// straight from global (L2-resident).
// ---------------------------------------------------------------------------
#define EW 4
__global__ void edge_kernel(const float* __restrict__ ea,
                            const int* __restrict__ eix,        // [2][E]
                            const float* __restrict__ h,        // [N][128]
                            const bf16* __restrict__ w1sw,      // swz 64x64
                            const float* __restrict__ b1,
                            const bf16* __restrict__ w2sw,      // swz 64x64
                            const float* __restrict__ b2,
                            const bf16* __restrict__ n2fsw,     // swz 128x64
                            const float* __restrict__ nb,
                            float* __restrict__ agg) {          // [N][64]
  __shared__ alignas(16) bf16 tile[EW][16 * 128];
  __shared__ int   rowix[EW][16];
  __shared__ int   colix[EW][16];
  __shared__ float eatt[EW][16];

  const int tid = threadIdx.x, lane = tid & 31, w = tid >> 5;
  const int e0 = (blockIdx.x * EW + w) * 16;
  const int n = lane & 15, hlf = lane >> 4;

  if (lane < 16) {
    int e = e0 + lane;
    bool ok = e < NEDGE;
    rowix[w][lane] = ok ? eix[e] : 0;
    colix[w][lane] = ok ? eix[NEDGE + e] : 0;
    eatt[w][lane]  = ok ? ea[e] : 0.0f;
  }
  __syncthreads();

  // Phase 1: RBF 16x64 (K padded 50->64), packed v2bf stores
  for (int ii = lane; ii < 512; ii += 32) {
    int m = ii >> 5, p = ii & 31;
    int k = p * 2;
    float d = eatt[w][m];
    float t0 = (d - (float)k * RBF_W) / RBF_W;
    float t1 = (d - (float)(k + 1) * RBF_W) / RBF_W;
    v2bf pk;
    pk[0] = (k     < NGAUSS) ? tobf(__expf(-0.5f * t0 * t0)) : tobf(0.0f);
    pk[1] = (k + 1 < NGAUSS) ? tobf(__expf(-0.5f * t1 * t1)) : tobf(0.0f);
    *(v2bf*)&tile[w][m * 64 + k] = pk;
  }
  __syncthreads();

  // Phase 2: GEMM1 (16x64)@(64x64)
  v16bf a0 = frag_a(tile[w], 64, 0, lane);
  v16bf a1 = frag_a(tile[w], 64, 32, lane);
  v8f acc[4];
#pragma unroll
  for (int t = 0; t < 4; ++t) {
    v8f c = {};
    c = wmma_bf(a0, frag_bg(w1sw, 0 * 4 + t, lane), c);
    c = wmma_bf(a1, frag_bg(w1sw, 1 * 4 + t, lane), c);
    acc[t] = c;
  }
  __syncthreads();
#pragma unroll
  for (int t = 0; t < 4; ++t) {
    float bb = b1[t * 16 + n];
#pragma unroll
    for (int v = 0; v < 8; ++v)
      tile[w][(v + 8 * hlf) * 64 + t * 16 + n] = tobf(silu_f(acc[t][v] + bb));
  }
  __syncthreads();

  // Phase 3: GEMM2 (16x64)@(64x64) -> W fragments (kept in registers)
  a0 = frag_a(tile[w], 64, 0, lane);
  a1 = frag_a(tile[w], 64, 32, lane);
  v8f Wacc[4];
#pragma unroll
  for (int t = 0; t < 4; ++t) {
    v8f c = {};
    c = wmma_bf(a0, frag_bg(w2sw, 0 * 4 + t, lane), c);
    c = wmma_bf(a1, frag_bg(w2sw, 1 * 4 + t, lane), c);
    float bb = b2[t * 16 + n];
#pragma unroll
    for (int v = 0; v < 8; ++v) c[v] += bb;
    Wacc[t] = c;
  }
  __syncthreads();

  // Phase 4: gather h[col] -> 16x128 bf16 tile, v4f loads / v4bf stores
  for (int ii = lane; ii < 512; ii += 32) {
    int m = ii >> 5, c4 = (ii & 31) << 2;
    v4f f = *(const v4f*)&h[(size_t)colix[w][m] * HID + c4];
    v4bf o;
    o[0] = tobf(f[0]); o[1] = tobf(f[1]); o[2] = tobf(f[2]); o[3] = tobf(f[3]);
    *(v4bf*)&tile[w][m * 128 + c4] = o;
  }
  __syncthreads();

  // Phase 5: GEMM3 (16x128)@(128x64) -> hj; msg = hj*W; atomic scatter
  v16bf g0 = frag_a(tile[w], 128, 0, lane);
  v16bf g1 = frag_a(tile[w], 128, 32, lane);
  v16bf g2 = frag_a(tile[w], 128, 64, lane);
  v16bf g3 = frag_a(tile[w], 128, 96, lane);
  v8f hj[4];
#pragma unroll
  for (int t = 0; t < 4; ++t) {
    v8f c = {};
    c = wmma_bf(g0, frag_bg(n2fsw, 0 * 4 + t, lane), c);
    c = wmma_bf(g1, frag_bg(n2fsw, 1 * 4 + t, lane), c);
    c = wmma_bf(g2, frag_bg(n2fsw, 2 * 4 + t, lane), c);
    c = wmma_bf(g3, frag_bg(n2fsw, 3 * 4 + t, lane), c);
    float bb = nb[t * 16 + n];
#pragma unroll
    for (int v = 0; v < 8; ++v) c[v] = (c[v] + bb) * Wacc[t][v];
    hj[t] = c;
  }

  int eCap = NEDGE - e0;
  if (eCap >= 16) {            // uniform fast path (E % 64 == 0 -> always)
#pragma unroll
    for (int t = 0; t < 4; ++t)
#pragma unroll
      for (int v = 0; v < 8; ++v) {
        int m = v + 8 * hlf;
        atomic_addf(&agg[(size_t)rowix[w][m] * FLT + t * 16 + n], hj[t][v]);
      }
  } else {
#pragma unroll
    for (int t = 0; t < 4; ++t)
#pragma unroll
      for (int v = 0; v < 8; ++v) {
        int m = v + 8 * hlf;
        if (m < eCap)
          atomic_addf(&agg[(size_t)rowix[w][m] * FLT + t * 16 + n], hj[t][v]);
      }
  }
}

// ---------------------------------------------------------------------------
// pooling
// ---------------------------------------------------------------------------
__global__ void pool_sum_kernel(const float* __restrict__ h, const int* __restrict__ batch,
                                float* __restrict__ pooled, float* __restrict__ counts) {
  int i = blockIdx.x * blockDim.x + threadIdx.x;
  if (i >= NNODE * HID) return;
  int nn = i >> 7, c = i & 127;
  int g = batch[nn];
  atomic_addf(&pooled[(size_t)g * HID + c], h[i]);
  if (c == 0) atomic_addf(&counts[g], 1.0f);
}

__global__ void pool_div_kernel(float* __restrict__ pooled, const float* __restrict__ counts) {
  int i = blockIdx.x * blockDim.x + threadIdx.x;
  if (i >= NGRAPH * HID) return;
  pooled[i] *= __builtin_amdgcn_rcpf(fmaxf(counts[i >> 7], 1.0f));
}

// ---------------------------------------------------------------------------
extern "C" void kernel_launch(void* const* d_in, const int* in_sizes, int n_in,
                              void* d_out, int out_size, void* d_ws, size_t ws_size,
                              hipStream_t stream) {
  const float* z         = (const float*)d_in[0];
  // d_in[1] = pos (unused by reference)
  const int*   edge_idx  = (const int*)d_in[2];
  const float* edge_attr = (const float*)d_in[3];
  const int*   batch     = (const int*)d_in[4];
  const float* embed_w1  = (const float*)d_in[5];
  const float* embed_b1  = (const float*)d_in[6];
  const float* embed_w2  = (const float*)d_in[7];
  const float* embed_b2  = (const float*)d_in[8];
  const float* filt_w1   = (const float*)d_in[9];
  const float* filt_b1   = (const float*)d_in[10];
  const float* filt_w2   = (const float*)d_in[11];
  const float* filt_b2   = (const float*)d_in[12];
  const float* n2f_w     = (const float*)d_in[13];
  const float* n2f_b     = (const float*)d_in[14];
  const float* f2n_w1    = (const float*)d_in[15];
  const float* f2n_b1    = (const float*)d_in[16];
  const float* f2n_w2    = (const float*)d_in[17];
  const float* f2n_b2    = (const float*)d_in[18];
  const float* ro_w1     = (const float*)d_in[19];
  const float* ro_b1     = (const float*)d_in[20];
  const float* ro_w2     = (const float*)d_in[21];
  const float* ro_b2     = (const float*)d_in[22];
  float* out = (float*)d_out;

  // workspace carve-up (256B aligned)
  char* base = (char*)d_ws;
  size_t off = 0;
  auto carve = [&](size_t bytes) { char* p = base + off; off = (off + bytes + 255) & ~(size_t)255; return p; };
  bf16* wb_emb2  = (bf16*)carve(HID * HID * 2);
  bf16* wb_fw1   = (bf16*)carve(NLAYER * 64 * FLT * 2);   // K padded 50->64
  bf16* wb_fw2   = (bf16*)carve(NLAYER * FLT * FLT * 2);
  bf16* wb_n2f   = (bf16*)carve(NLAYER * HID * FLT * 2);
  bf16* wb_f2n1  = (bf16*)carve(NLAYER * FLT * HID * 2);
  bf16* wb_f2n2  = (bf16*)carve(NLAYER * HID * HID * 2);
  bf16* wb_ro1   = (bf16*)carve(HID * HID * 2);
  bf16* wb_ro2   = (bf16*)carve(HID * HID * 2);
  float* t1      = (float*)carve((size_t)NNODE * HID * 4);
  float* hbuf    = (float*)carve((size_t)NNODE * HID * 4);
  float* agg     = (float*)carve((size_t)NNODE * FLT * 4);
  float* pooled  = (float*)carve((size_t)NGRAPH * HID * 4);
  float* counts  = (float*)carve((size_t)NGRAPH * 4);
  float* t3      = (float*)carve((size_t)NGRAPH * HID * 4);
  (void)ws_size; (void)in_sizes; (void)n_in; (void)out_size;

  // swizzled weight conversion (per-matrix; stacked weights per layer)
  auto cvt = [&](const float* s, bf16* d, int srcRows, int K, int cols) {
    int tot = K * cols;
    cvt_swz_kernel<<<(tot + 255) / 256, 256, 0, stream>>>(s, d, srcRows, K, cols);
  };
  cvt(embed_w2, wb_emb2, HID, HID, HID);
  cvt(ro_w1, wb_ro1, HID, HID, HID);
  cvt(ro_w2, wb_ro2, HID, HID, HID);
  for (int l = 0; l < NLAYER; ++l) {
    cvt(filt_w1 + (size_t)l * NGAUSS * FLT, wb_fw1 + (size_t)l * 64 * FLT, NGAUSS, 64, FLT);
    cvt(filt_w2 + (size_t)l * FLT * FLT,   wb_fw2 + (size_t)l * FLT * FLT, FLT, FLT, FLT);
    cvt(n2f_w  + (size_t)l * HID * FLT,    wb_n2f + (size_t)l * HID * FLT, HID, HID, FLT);
    cvt(f2n_w1 + (size_t)l * FLT * HID,    wb_f2n1 + (size_t)l * FLT * HID, FLT, FLT, HID);
    cvt(f2n_w2 + (size_t)l * HID * HID,    wb_f2n2 + (size_t)l * HID * HID, HID, HID, HID);
  }

  // node embedding: t1 = silu(z@W1+b1); h = t1@W2+b2
  emb1_kernel<<<(NNODE * HID + 255) / 256, 256, 0, stream>>>(z, embed_w1, embed_b1, t1);
  int nodeBlocks = (NNODE + 63) / 64;
  gemm_node_kernel<HID><<<nodeBlocks, 128, 0, stream>>>(t1, NNODE, wb_emb2, embed_b2,
                                                        hbuf, /*act=*/0, /*addTo=*/0);

  // interaction layers
  int edgeBlocks = (NEDGE + (EW * 16) - 1) / (EW * 16);
  for (int l = 0; l < NLAYER; ++l) {
    hipMemsetAsync(agg, 0, (size_t)NNODE * FLT * 4, stream);
    edge_kernel<<<edgeBlocks, 128, 0, stream>>>(
        edge_attr, edge_idx, hbuf,
        wb_fw1 + (size_t)l * 64 * FLT,  filt_b1 + (size_t)l * FLT,
        wb_fw2 + (size_t)l * FLT * FLT, filt_b2 + (size_t)l * FLT,
        wb_n2f + (size_t)l * HID * FLT, n2f_b + (size_t)l * FLT,
        agg);
    gemm_node_kernel<FLT><<<nodeBlocks, 128, 0, stream>>>(agg, NNODE,
        wb_f2n1 + (size_t)l * FLT * HID, f2n_b1 + (size_t)l * HID,
        t1, /*act=*/1, /*addTo=*/0);
    gemm_node_kernel<HID><<<nodeBlocks, 128, 0, stream>>>(t1, NNODE,
        wb_f2n2 + (size_t)l * HID * HID, f2n_b2 + (size_t)l * HID,
        hbuf, /*act=*/0, /*addTo=*/1);
  }

  // global mean pool + readout
  hipMemsetAsync(pooled, 0, (size_t)NGRAPH * HID * 4, stream);
  hipMemsetAsync(counts, 0, (size_t)NGRAPH * 4, stream);
  pool_sum_kernel<<<(NNODE * HID + 255) / 256, 256, 0, stream>>>(hbuf, batch, pooled, counts);
  pool_div_kernel<<<(NGRAPH * HID + 255) / 256, 256, 0, stream>>>(pooled, counts);

  int roBlocks = (NGRAPH + 63) / 64;
  gemm_node_kernel<HID><<<roBlocks, 128, 0, stream>>>(pooled, NGRAPH, wb_ro1, ro_b1,
                                                      t3, /*act=*/1, /*addTo=*/0);
  gemm_node_kernel<HID><<<roBlocks, 128, 0, stream>>>(t3, NGRAPH, wb_ro2, ro_b2,
                                                      out, /*act=*/0, /*addTo=*/0);
}
